// Attention_7567732375671
// MI455X (gfx1250) — matrix operand order repaired and checked by reading, not compiled, not run on hardware
//
#include <hip/hip_runtime.h>
#include <hip/hip_bf16.h>

// MHA forward for B=4, N=1024, D=1024, H=16, DH=64 on gfx1250 (wave32, WMMA).
// All matrix math via v_wmma_f32_16x16x32_bf16 (bf16 in, f32 accumulate).
// Outputs: d_out[0 .. 4M)   = out [B,N,D] f32
//          d_out[4M .. 68M) = attn [B,H,N,N] f32
// Workspace (needs ~40.1 MB):
//   [0,8M)    Q  bf16 [B,H,N,64]
//   [8M,16M)  K  bf16 [B,H,N,64]
//   [16M,24M) V^T bf16 [B,H,64,N]
//   [24M,40M) O  f32  [B,N,D]
//   [40M,+16KB) mfull f32 [B,N]
//
// Register-reuse design: attn [268MB] is the only tensor larger than the
// 192MB L2, so pv_gemm reads each attn fragment exactly once (wave computes
// the full 16x64 output band). qk_logits reuses the Q fragment across 4
// j-tiles; the projection GEMMs reuse the A fragment across 2 n-tiles with
// a 32-column LDS-staged weight tile.

#define SCALE_ 0.03125f           // D^-0.5 = 1/32
#define FMAXV  3.402823466e38f

typedef __bf16 bf16_t;
typedef __attribute__((ext_vector_type(16))) __bf16 v16bf;
typedef __attribute__((ext_vector_type(8)))  float  v8f;

// ---- WMMA fragment loaders (layouts per CDNA5 ISA §7.12.2) ----
// A 16x32 bf16: lane&15 = row M; lane>>4 selects K-halves {0..7,16..23} vs {8..15,24..31}
static __device__ __forceinline__ v16bf frag_a_f32(const float* __restrict__ base,
                                                   int stride, int kc, int lane) {
  const int row = lane & 15, hs = lane >> 4;
  const float* p = base + (size_t)row * stride + kc + hs * 8;
  v16bf a;
#pragma unroll
  for (int e = 0; e < 8; ++e) a[e] = (bf16_t)p[e];          // K = hs*8 + e
#pragma unroll
  for (int e = 0; e < 8; ++e) a[8 + e] = (bf16_t)p[16 + e]; // K = 16 + hs*8 + e
  return a;
}

static __device__ __forceinline__ v16bf frag_a_bf16(const bf16_t* __restrict__ base,
                                                    int stride, int kc, int lane) {
  const int row = lane & 15, hs = lane >> 4;
  const bf16_t* p = base + (size_t)row * stride + kc + hs * 8;
  v16bf a;
#pragma unroll
  for (int e = 0; e < 8; ++e) a[e] = p[e];
#pragma unroll
  for (int e = 0; e < 8; ++e) a[8 + e] = p[16 + e];
  return a;
}

// B 32x16 bf16: lane&15 = column N; lane>>4 selects K-halves {0..7,16..23} vs {8..15,24..31}.
// Column n of B is row n of a row-major [rows, stride] array (NT-GEMM friendly).
static __device__ __forceinline__ v16bf frag_b_bf16(const bf16_t* __restrict__ base,
                                                    size_t stride, int kc, int lane) {
  const int col = lane & 15, hs = lane >> 4;
  const bf16_t* p = base + (size_t)col * stride + kc + hs * 8;
  v16bf b;
#pragma unroll
  for (int e = 0; e < 8; ++e) b[e] = p[e];
#pragma unroll
  for (int e = 0; e < 8; ++e) b[8 + e] = p[16 + e];
  return b;
}

// ---------------- Kernel 1: qkv = x @ Wqkv + bqkv, scatter to Q/K/V^T ----------------
// Block = 128 threads (4 waves); block tile = 64(M)x32(N); wave tile = 16x32 (2 acc).
// W tile (32K x 32N) staged in LDS as bf16, column-major [col][32] for contiguous B-frags.
__global__ __launch_bounds__(128) void qkv_gemm(const float* __restrict__ x,
                                                const float* __restrict__ W,
                                                const float* __restrict__ bias,
                                                bf16_t* __restrict__ qb,
                                                bf16_t* __restrict__ kb,
                                                bf16_t* __restrict__ vtb) {
  __shared__ bf16_t ldsB[32 * 32];
  const int tid = threadIdx.x, lane = tid & 31, wave = tid >> 5;
  const int m0 = blockIdx.x * 64, n0 = blockIdx.y * 32;
  const float* abase = x + (size_t)(m0 + wave * 16) * 1024;
  v8f c0 = {}, c1 = {};
  for (int k0 = 0; k0 < 1024; k0 += 32) {
    // cooperative stage: 128 threads x 8 elems = 1024 = 32x32 tile, transposed into LDS
    {
      const int col = tid >> 2, kk = (tid & 3) * 8;
      const float* wp = W + (size_t)(k0 + kk) * 3072 + n0 + col;
#pragma unroll
      for (int j = 0; j < 8; ++j) ldsB[col * 32 + kk + j] = (bf16_t)wp[(size_t)j * 3072];
    }
    if (k0 + 32 < 1024)  // gfx1250 global_prefetch_b8 of next W tile
      __builtin_prefetch(W + (size_t)(k0 + 32) * 3072 + n0 + (tid >> 2), 0, 1);
    __syncthreads();
    v16bf a  = frag_a_f32(abase, 1024, k0, lane);
    v16bf b0 = frag_b_bf16(ldsB,      32, 0, lane);
    v16bf b1 = frag_b_bf16(ldsB + 16 * 32, 32, 0, lane);
    c0 = __builtin_amdgcn_wmma_f32_16x16x32_bf16(false, a, false, b0, (short)0, c0, false, false);
    c1 = __builtin_amdgcn_wmma_f32_16x16x32_bf16(false, a, false, b1, (short)0, c1, false, false);
    __syncthreads();
  }
  const int hs = lane >> 4, col = lane & 15;
#pragma unroll
  for (int nt = 0; nt < 2; ++nt) {
    const v8f& c = nt ? c1 : c0;
    const int gn = n0 + nt * 16 + col;
    const float bv = bias[gn];
#pragma unroll
    for (int vi = 0; vi < 8; ++vi) {
      const int gm = m0 + wave * 16 + vi + 8 * hs;  // C layout: M = vgpr + 8*(lane>=16)
      const int b = gm >> 10, n = gm & 1023;
      const float val = c[vi] + bv;
      if (gn < 1024) {                 // Q head-split [B,H,N,64]
        const int h = gn >> 6, d = gn & 63;
        qb[((size_t)(b * 16 + h) * 1024 + n) * 64 + d] = (bf16_t)val;
      } else if (gn < 2048) {          // K head-split [B,H,N,64]
        const int g = gn - 1024, h = g >> 6, d = g & 63;
        kb[((size_t)(b * 16 + h) * 1024 + n) * 64 + d] = (bf16_t)val;
      } else {                         // V transposed [B,H,64,N]
        const int g = gn - 2048, h = g >> 6, d = g & 63;
        vtb[((size_t)(b * 16 + h) * 64 + d) * 1024 + n] = (bf16_t)val;
      }
    }
  }
}

// ---------------- Kernel 2: mfull[b,i] = (i==0) | mask[b,i-1] ----------------
__global__ __launch_bounds__(256) void build_mask(const unsigned char* __restrict__ mask,
                                                  float* __restrict__ mfull) {
  const int idx = blockIdx.x * 256 + threadIdx.x;  // B*N = 4096
  if (idx >= 4 * 1024) return;
  const int b = idx >> 10, i = idx & 1023;
  mfull[idx] = (i == 0) ? 1.f : (mask[b * 1023 + i - 1] ? 1.f : 0.f);
}

// ---------------- Kernel 3: logits = scale * Q K^T, masked -> attn region ----------------
// 256 threads = 8 waves; wave handles one 16x64 band (Q frag loaded once, 8 wmma).
__global__ __launch_bounds__(256) void qk_logits(const bf16_t* __restrict__ qb,
                                                 const bf16_t* __restrict__ kb,
                                                 const float* __restrict__ mfull,
                                                 float* __restrict__ attn) {
  const int lane = threadIdx.x & 31, wave = threadIdx.x >> 5;
  const int bh = blockIdx.z;
  const int i0 = (blockIdx.y * 8 + wave) * 16;
  const int j0 = blockIdx.x * 64;
  const int b = bh >> 4;
  const bf16_t* qbase = qb + ((size_t)bh * 1024 + i0) * 64;
  const bf16_t* kbase = kb + ((size_t)bh * 1024 + j0) * 64;  // row j of K = column j of K^T
  const v16bf a0 = frag_a_bf16(qbase, 64, 0, lane);
  const v16bf a1 = frag_a_bf16(qbase, 64, 32, lane);
  v8f acc[4] = {{}, {}, {}, {}};
#pragma unroll
  for (int jt = 0; jt < 4; ++jt) {
    v16bf b0 = frag_b_bf16(kbase + (size_t)(jt * 16) * 64, 64, 0, lane);
    v16bf b1 = frag_b_bf16(kbase + (size_t)(jt * 16) * 64, 64, 32, lane);
    acc[jt] = __builtin_amdgcn_wmma_f32_16x16x32_bf16(false, a0, false, b0, (short)0, acc[jt], false, false);
    acc[jt] = __builtin_amdgcn_wmma_f32_16x16x32_bf16(false, a1, false, b1, (short)0, acc[jt], false, false);
  }
  const int hs = lane >> 4, col = lane & 15;
#pragma unroll
  for (int jt = 0; jt < 4; ++jt) {
    const int j = j0 + jt * 16 + col;
    const float mj = mfull[b * 1024 + j];
#pragma unroll
    for (int vi = 0; vi < 8; ++vi) {
      const int i = i0 + vi + 8 * hs;
      const float mi = mfull[b * 1024 + i];
      const float val = (mi * mj != 0.f) ? acc[jt][vi] * SCALE_ : -FMAXV;
      attn[((size_t)bh * 1024 + i) * 1024 + j] = val;
    }
  }
}

// ---------------- Kernel 4: row softmax over attn (in place) ----------------
__global__ __launch_bounds__(256) void softmax_rows(float* __restrict__ attn) {
  __shared__ float red[256];
  float* p = attn + (size_t)blockIdx.x * 1024;
  const int t = threadIdx.x;
  float4 v = ((float4*)p)[t];
  float mx = fmaxf(fmaxf(v.x, v.y), fmaxf(v.z, v.w));
  red[t] = mx; __syncthreads();
#pragma unroll
  for (int s = 128; s > 0; s >>= 1) { if (t < s) red[t] = fmaxf(red[t], red[t + s]); __syncthreads(); }
  mx = red[0]; __syncthreads();
  v.x = __expf(v.x - mx); v.y = __expf(v.y - mx);
  v.z = __expf(v.z - mx); v.w = __expf(v.w - mx);
  red[t] = v.x + v.y + v.z + v.w; __syncthreads();
#pragma unroll
  for (int s = 128; s > 0; s >>= 1) { if (t < s) red[t] += red[t + s]; __syncthreads(); }
  const float inv = 1.f / red[0];
  v.x *= inv; v.y *= inv; v.z *= inv; v.w *= inv;
  ((float4*)p)[t] = v;
}

// ---------------- Kernel 5: O = attn @ V (V^T layout), merged to [B,N,D] ----------------
// Wave computes full 16x64 output band: the HBM-resident attn fragment is loaded
// exactly once and reused across all 4 d-tiles (4 wmma per K-step).
__global__ __launch_bounds__(256) void pv_gemm(const float* __restrict__ attn,
                                               const bf16_t* __restrict__ vtb,
                                               float* __restrict__ O) {
  const int lane = threadIdx.x & 31, wave = threadIdx.x >> 5;
  const int bh = blockIdx.y;
  const int i0 = (blockIdx.x * 8 + wave) * 16;
  const float* abase = attn + ((size_t)bh * 1024 + i0) * 1024;
  const bf16_t* vbase = vtb + (size_t)bh * 64 * 1024;  // rows d of V^T = columns d of V
  v8f acc[4] = {{}, {}, {}, {}};
  for (int kc = 0; kc < 1024; kc += 32) {
    v16bf a = frag_a_f32(abase, 1024, kc, lane);
#pragma unroll
    for (int dt = 0; dt < 4; ++dt) {
      v16bf bfr = frag_b_bf16(vbase + (size_t)(dt * 16) * 1024, 1024, kc, lane);
      acc[dt] = __builtin_amdgcn_wmma_f32_16x16x32_bf16(false, a, false, bfr, (short)0, acc[dt], false, false);
    }
  }
  const int hs = lane >> 4, col = lane & 15;
  const int b = bh >> 4, h = bh & 15;
#pragma unroll
  for (int dt = 0; dt < 4; ++dt) {
    const int d = dt * 16 + col;
#pragma unroll
    for (int vi = 0; vi < 8; ++vi) {
      const int i = i0 + vi + 8 * hs;
      O[((size_t)b * 1024 + i) * 1024 + h * 64 + d] = acc[dt][vi];  // head-merge transpose
    }
  }
}

// ---------------- Kernel 6: out = O @ Wout + bout ----------------
__global__ __launch_bounds__(128) void out_gemm(const float* __restrict__ O,
                                                const float* __restrict__ W,
                                                const float* __restrict__ bias,
                                                float* __restrict__ out) {
  __shared__ bf16_t ldsB[32 * 32];
  const int tid = threadIdx.x, lane = tid & 31, wave = tid >> 5;
  const int m0 = blockIdx.x * 64, n0 = blockIdx.y * 32;
  const float* abase = O + (size_t)(m0 + wave * 16) * 1024;
  v8f c0 = {}, c1 = {};
  for (int k0 = 0; k0 < 1024; k0 += 32) {
    {
      const int col = tid >> 2, kk = (tid & 3) * 8;
      const float* wp = W + (size_t)(k0 + kk) * 1024 + n0 + col;
#pragma unroll
      for (int j = 0; j < 8; ++j) ldsB[col * 32 + kk + j] = (bf16_t)wp[(size_t)j * 1024];
    }
    if (k0 + 32 < 1024)
      __builtin_prefetch(W + (size_t)(k0 + 32) * 1024 + n0 + (tid >> 2), 0, 1);
    __syncthreads();
    v16bf a  = frag_a_f32(abase, 1024, k0, lane);
    v16bf b0 = frag_b_bf16(ldsB,      32, 0, lane);
    v16bf b1 = frag_b_bf16(ldsB + 16 * 32, 32, 0, lane);
    c0 = __builtin_amdgcn_wmma_f32_16x16x32_bf16(false, a, false, b0, (short)0, c0, false, false);
    c1 = __builtin_amdgcn_wmma_f32_16x16x32_bf16(false, a, false, b1, (short)0, c1, false, false);
    __syncthreads();
  }
  const int hs = lane >> 4, col = lane & 15;
#pragma unroll
  for (int nt = 0; nt < 2; ++nt) {
    const v8f& c = nt ? c1 : c0;
    const int gn = n0 + nt * 16 + col;
    const float bv = bias[gn];
#pragma unroll
    for (int vi = 0; vi < 8; ++vi) {
      const int gm = m0 + wave * 16 + vi + 8 * hs;
      out[(size_t)gm * 1024 + gn] = c[vi] + bv;
    }
  }
}

extern "C" void kernel_launch(void* const* d_in, const int* in_sizes, int n_in,
                              void* d_out, int out_size, void* d_ws, size_t ws_size,
                              hipStream_t stream) {
  const float* x          = (const float*)d_in[0];
  const unsigned char* mk = (const unsigned char*)d_in[1];  // bool mask [B, N-1]
  const float* Wqkv       = (const float*)d_in[2];
  const float* bqkv       = (const float*)d_in[3];
  const float* Wout       = (const float*)d_in[4];
  const float* bout       = (const float*)d_in[5];

  float* out  = (float*)d_out;
  float* attn = out + (size_t)4 * 1024 * 1024;   // second tuple output

  char* ws = (char*)d_ws;                        // needs >= 40MB + 16KB
  bf16_t* qb    = (bf16_t*)(ws);
  bf16_t* kb    = (bf16_t*)(ws + (((size_t)8)  << 20));
  bf16_t* vtb   = (bf16_t*)(ws + (((size_t)16) << 20));
  float*  Obuf  = (float*) (ws + (((size_t)24) << 20));
  float*  mfull = (float*) (ws + (((size_t)40) << 20));

  build_mask<<<16, 256, 0, stream>>>(mk, mfull);
  qkv_gemm<<<dim3(64, 96), 128, 0, stream>>>(x, Wqkv, bqkv, qb, kb, vtb);
  qk_logits<<<dim3(16, 8, 64), 256, 0, stream>>>(qb, kb, mfull, attn);
  softmax_rows<<<dim3(4 * 16 * 1024), 256, 0, stream>>>(attn);
  pv_gemm<<<dim3(8, 64), 256, 0, stream>>>(attn, vtb, Obuf);
  out_gemm<<<dim3(64, 32), 128, 0, stream>>>(Obuf, Wout, bout, out);
}
